// cate_mixture_model_78030965833938
// MI455X (gfx1250) — matrix operand order repaired and checked by reading, not compiled, not run on hardware
//
#include <hip/hip_runtime.h>
#include <stdint.h>

typedef __attribute__((ext_vector_type(16))) _Float16     v16h;
typedef __attribute__((ext_vector_type(8)))  float        v8f;
typedef __attribute__((ext_vector_type(4)))  unsigned int v4u;
typedef __attribute__((ext_vector_type(4)))  int          v4i;
typedef __attribute__((ext_vector_type(8)))  int          v8i;

#define M_MIX 16
#define D_VIS 128
#define C_CAT 50
#define C_PAD 64
#define B_TOT 32768

// B-image: per (d, ktile, part) one 1024-byte page holding the WMMA 16-bit
// B operand (K=32 x N=16, f16) in register-striped order:
//   lane = m + 16*(k>=16), VGPR j packs K rows (2j, 2j+1) of the lane's half.
// img byte size = 128 * 2 * 2 * 1024 = 524288.
#define IMG_BYTES (D_VIS * 2 * 2 * 1024)

__device__ __forceinline__ uint32_t img_off(int d, int c, int m, int p) {
    int kt  = c >> 5;          // ktile
    int k   = c & 31;          // k within tile
    int hh  = (k >> 4) & 1;    // lane half
    int j   = (k & 15) >> 1;   // packed VGPR index
    int odd = k & 1;
    return (uint32_t)((((d * 2 + kt) * 2 + p) << 10) +
                      ((m + 16 * hh) << 5) + (j << 2) + (odd << 1));
}

// ---------------------------------------------------------------------------
// Kernel 1: log-softmax of mixture weights and categorical logits; emit the
// pre-swizzled hi/lo f16 B-image (hi = f16(logp), lo = f16(logp - hi)).
// ---------------------------------------------------------------------------
__global__ void prep_kernel(const float* __restrict__ mw,
                            const float* __restrict__ cate,
                            unsigned char* __restrict__ img,
                            float* __restrict__ logw) {
    int tid = blockIdx.x * blockDim.x + threadIdx.x;
    if (tid == 0) {
        float mx = mw[0];
        for (int m = 1; m < M_MIX; ++m) mx = fmaxf(mx, mw[m]);
        float s = 0.0f;
        for (int m = 0; m < M_MIX; ++m) s += expf(mw[m] - mx);
        float lse = mx + logf(s);
        for (int m = 0; m < M_MIX; ++m) logw[m] = mw[m] - lse;
    }
    if (tid < M_MIX * D_VIS) {
        int m = tid / D_VIS;
        int d = tid - m * D_VIS;
        const float* row = cate + ((size_t)m * D_VIS + d) * C_CAT;
        float mx = row[0];
        for (int c = 1; c < C_CAT; ++c) mx = fmaxf(mx, row[c]);
        float s = 0.0f;
        for (int c = 0; c < C_CAT; ++c) s += expf(row[c] - mx);
        float lse = mx + logf(s);
        for (int c = 0; c < C_PAD; ++c) {
            _Float16 hi = (_Float16)0.0f, lo = (_Float16)0.0f;
            if (c < C_CAT) {
                float lp = row[c] - lse;
                hi = (_Float16)lp;
                lo = (_Float16)(lp - (float)hi);
            }
            *(_Float16*)(img + img_off(d, c, m, 0)) = hi;
            *(_Float16*)(img + img_off(d, c, m, 1)) = lo;
        }
    }
}

// ---------------------------------------------------------------------------
// TDM staging of one contiguous 4096-byte slice (global -> LDS).
// ---------------------------------------------------------------------------
#if defined(__has_builtin)
#if __has_builtin(__builtin_amdgcn_tensor_load_to_lds)
#define HAVE_TDM 1
#endif
#endif

__device__ __forceinline__ void stage_slice(const unsigned char* __restrict__ gsrc,
                                            unsigned char* __restrict__ ldst,
                                            int lane) {
#ifdef HAVE_TDM
    (void)lane;
    const uint64_t ga = (uint64_t)(uintptr_t)gsrc;
    const uint32_t la = (uint32_t)(uintptr_t)ldst;   // low 32 bits = LDS byte addr
    const uint32_t ne = 4096u >> 3;                  // 512 elements of 8 bytes
    // D# group 0: count=1 | lds_addr | global_addr[56:0] | type=2
    v4u g0 = { 1u, la, (uint32_t)ga,
               (((uint32_t)(ga >> 32)) & 0x01FFFFFFu) | (2u << 30) };
    // D# group 1: data_size=3 (8B), tensor_dim0=tile_dim0=512, dim1=1, stride0=512
    v8i g1 = { (int)(3u << 16),          // wg_mask=0, data_size=8B
               (int)(ne << 16),          // tensor_dim0[15:0] @ bits 63:48
               (int)(1u << 16),          // tensor_dim0[31:16]=0, tensor_dim1=1
               (int)(ne << 16),          // tile_dim0 @ bits 127:112
               (int)1,                   // tile_dim1=1, tile_dim2=0
               (int)ne,                  // tensor_dim0_stride[31:0]
               0, 0 };
    v4i z4 = { 0, 0, 0, 0 };
#if defined(__clang_major__) && (__clang_major__ >= 23)
    v8i z8 = { 0, 0, 0, 0, 0, 0, 0, 0 };
    __builtin_amdgcn_tensor_load_to_lds(g0, g1, z4, z4, z8, 0);
#else
    __builtin_amdgcn_tensor_load_to_lds(g0, g1, z4, z4, 0);
#endif
#else
    // Fallback: cooperative wave copy, 128 B per lane.
    const uint4* s = (const uint4*)(gsrc) + lane;
    uint4*       t = (uint4*)(ldst) + lane;
#pragma unroll
    for (int q = 0; q < 8; ++q) t[q * 32] = s[q * 32];
#endif
}

__device__ __forceinline__ void stage_wait() {
#ifdef HAVE_TDM
    __builtin_amdgcn_s_wait_tensorcnt(0);
#endif
}

// ---------------------------------------------------------------------------
// Kernel 2: per wave, a 16-row batch tile; D-loop over 16 chunks of 8 d's,
// double-buffered through LDS by the TDM; 4 WMMAs per d (2 ktiles x hi/lo).
// ---------------------------------------------------------------------------
__global__ void __launch_bounds__(256) mix_kernel(const int* __restrict__ x,
                                                  const unsigned char* __restrict__ img,
                                                  const float* __restrict__ logw,
                                                  float* __restrict__ out) {
    __shared__ unsigned char lds_buf[2 * 32768];     // double buffer, 8 d's each

    const int tid  = threadIdx.x;
    const int lane = tid & 31;
    const int hh   = (lane >> 4) & 1;                // lane half
    const int n    = lane & 15;                      // N index (mixture m / row id)
    const int wave = __builtin_amdgcn_readfirstlane(tid >> 5);   // 0..7
    const int b0   = (blockIdx.x * 8 + wave) * 16;   // batch tile base

    const float lw = logw[n];
    const int*  xrow = x + (size_t)(b0 + n) * D_VIS; // row for this lane's M=n

    v8f acc = {};

    // Prologue: stage chunk 0 (wave w stages d = w).
    stage_slice(img + (size_t)wave * 4096, &lds_buf[wave * 4096], lane);
    stage_wait();
    __syncthreads();

    for (int ch = 0; ch < 16; ++ch) {
        const int par = ch & 1;
        if (ch + 1 < 16) {
            const int dnext = (ch + 1) * 8 + wave;
            stage_slice(img + (size_t)dnext * 4096,
                        &lds_buf[(par ^ 1) * 32768 + wave * 4096], lane);
        }

        // x slice for this chunk: 8 contiguous ints (32 B, two b128 loads).
        int xs[8];
        {
            const int4* xp = (const int4*)(xrow + ch * 8);
#pragma unroll
            for (int q = 0; q < 2; ++q) {
                int4 v = xp[q];
                xs[4 * q + 0] = v.x; xs[4 * q + 1] = v.y;
                xs[4 * q + 2] = v.z; xs[4 * q + 3] = v.w;
            }
        }

        const unsigned char* base = &lds_buf[par * 32768];
#pragma unroll
        for (int dl = 0; dl < 8; ++dl) {
            const int xi = xs[dl];
            const unsigned char* dpage = base + dl * 4096 + lane * 32;
#pragma unroll
            for (int t = 0; t < 2; ++t) {
                // Build one-hot A row (M = n) for ktile t, with the 16-bit
                // 16x32 A in this code's contiguous k order: K=16h+2j (matches the B-image).
                v16h a;
#pragma unroll
                for (int j = 0; j < 8; ++j) {
                    int kb = 16 * hh + 2 * j;
                    int c0 = t * 32 + kb;
                    a[2 * j + 0] = (xi == c0)     ? (_Float16)1.0f : (_Float16)0.0f;
                    a[2 * j + 1] = (xi == c0 + 1) ? (_Float16)1.0f : (_Float16)0.0f;
                }
                v16h bhi = *(const v16h*)(dpage + t * 2048);
                v16h blo = *(const v16h*)(dpage + t * 2048 + 1024);
                acc = __builtin_amdgcn_wmma_f32_16x16x32_f16(
                        false, a, false, bhi, (short)0, acc, false, false);
                acc = __builtin_amdgcn_wmma_f32_16x16x32_f16(
                        false, a, false, blo, (short)0, acc, false, false);
            }
        }
        stage_wait();
        __syncthreads();
    }

    // logsumexp over mixtures: N dim is striped across 16-lane groups.
    // acc VGPR j holds D[row = j + 8*hh, m = n].
#pragma unroll
    for (int j = 0; j < 8; ++j) {
        float v  = acc[j] + lw;
        float mx = v;
        mx = fmaxf(mx, __shfl_xor(mx, 8, 32));
        mx = fmaxf(mx, __shfl_xor(mx, 4, 32));
        mx = fmaxf(mx, __shfl_xor(mx, 2, 32));
        mx = fmaxf(mx, __shfl_xor(mx, 1, 32));
        float e = expf(v - mx);
        e += __shfl_xor(e, 8, 32);
        e += __shfl_xor(e, 4, 32);
        e += __shfl_xor(e, 2, 32);
        e += __shfl_xor(e, 1, 32);
        float res = mx + logf(e);
        if (n == 0) out[b0 + j + 8 * hh] = res;
    }
}

// ---------------------------------------------------------------------------
extern "C" void kernel_launch(void* const* d_in, const int* in_sizes, int n_in,
                              void* d_out, int out_size, void* d_ws, size_t ws_size,
                              hipStream_t stream) {
    (void)in_sizes; (void)n_in; (void)out_size; (void)ws_size;
    const int*   x    = (const int*)d_in[0];
    const float* mw   = (const float*)d_in[1];
    const float* cate = (const float*)d_in[2];
    float*       out  = (float*)d_out;

    unsigned char* img  = (unsigned char*)d_ws;            // 524288 B
    float*         logw = (float*)(img + IMG_BYTES);       // +64 B

    prep_kernel<<<8, 256, 0, stream>>>(mw, cate, img, logw);
    mix_kernel<<<B_TOT / (8 * 16), 256, 0, stream>>>(x, img, logw, out);
}
